// QAttention_69114613730505
// MI455X (gfx1250) — 8-bit matrix operand order repaired and checked by reading, compiled, not run on hardware
//
#include <hip/hip_runtime.h>


typedef __attribute__((ext_vector_type(8)))  int          v8i;
typedef __attribute__((ext_vector_type(4)))  int          v4i;
typedef __attribute__((ext_vector_type(2)))  int          v2i;
typedef __attribute__((ext_vector_type(4)))  float        v4f;
typedef __attribute__((ext_vector_type(4)))  unsigned int v4u;
typedef __attribute__((ext_vector_type(2)))  unsigned int v2u;

#define BATCH 8
#define SEQ   2048
#define CDIM  64

__device__ __forceinline__ void sched_fence() {
#if __has_builtin(__builtin_amdgcn_sched_barrier)
  __builtin_amdgcn_sched_barrier(0);
#endif
}

__device__ __forceinline__ unsigned perm(unsigned a, unsigned b, unsigned sel) {
  return __builtin_amdgcn_perm(a, b, sel);
}
__device__ __forceinline__ int pack_byte1(int s0, int s1, int s2, int s3) {
  unsigned p01 = perm((unsigned)s1, (unsigned)s0, 0x00000501u);
  unsigned p23 = perm((unsigned)s3, (unsigned)s2, 0x00000501u);
  return (int)perm(p23, p01, 0x05040100u);
}
__device__ __forceinline__ int pack_byte0(int s0, int s1, int s2, int s3) {
  unsigned p01 = perm((unsigned)s1, (unsigned)s0, 0x00000400u);
  unsigned p23 = perm((unsigned)s3, (unsigned)s2, 0x00000400u);
  return (int)perm(p23, p01, 0x05040100u);
}
__device__ __forceinline__ int pack4f(v4f f) {
  return pack_byte0((int)f[0], (int)f[1], (int)f[2], (int)f[3]);
}
__device__ __forceinline__ v8i make_v8(v4u lo, v4u hi) {
  v4i l = __builtin_bit_cast(v4i, lo);
  v4i h = __builtin_bit_cast(v4i, hi);
  return __builtin_shufflevector(l, h, 0, 1, 2, 3, 4, 5, 6, 7);
}

__global__ __launch_bounds__(32) void qattn_proj_kernel(
    const float* __restrict__ q_in, const float* __restrict__ k_in,
    const float* __restrict__ v_in,
    const float* __restrict__ wq, const float* __restrict__ wk,
    const float* __restrict__ wv,
    unsigned char* __restrict__ Qbuf, unsigned char* __restrict__ Kbuf,
    unsigned char* __restrict__ VTbuf)
{
  const int lane = threadIdx.x;
  const int n16  = lane & 15;
  const int h    = lane >> 4;
  const int job  = blockIdx.y;
  const float* x = (job == 0) ? q_in : (job == 1) ? k_in : v_in;
  const float* w = (job == 0) ? wq   : (job == 1) ? wk   : wv;

  const int row0 = blockIdx.x * 16;

  v8i a;
  const float* xrow = x + (size_t)(row0 + n16) * CDIM;
  #pragma unroll
  for (int g = 0; g < 4; ++g) {
    const int f0 = (g >> 1) * 32 + h * 16 + (g & 1) * 8;
    a[2*g]   = pack4f(*(const v4f*)(xrow + f0));
    a[2*g+1] = pack4f(*(const v4f*)(xrow + f0 + 4));
  }

  v8i bm[4];
  #pragma unroll
  for (int dt = 0; dt < 4; ++dt) {
    const float* wrow = w + (size_t)(dt * 16 + n16) * CDIM;
    #pragma unroll
    for (int v = 0; v < 8; ++v) {
      const int ks = (v >> 2) * 32 + h * 16 + (v & 3) * 4;
      bm[dt][v] = pack4f(*(const v4f*)(wrow + ks));
    }
  }

  sched_fence();

  v8i st[4];
  #pragma unroll
  for (int dt = 0; dt < 4; ++dt) {
    v8i c = {};
    st[dt] = __builtin_amdgcn_wmma_i32_16x16x64_iu8(false, a, false, bm[dt],
                                                    c, false, false);
  }

  if (job < 2) {
    __shared__ __align__(16) unsigned char lds[16 * CDIM];
    #pragma unroll
    for (int dt = 0; dt < 4; ++dt)
      #pragma unroll
      for (int r = 0; r < 8; ++r)
        lds[(r + 8*h) * CDIM + dt * 16 + n16] =
            (unsigned char)(((unsigned)st[dt][r]) >> 8);
    __syncthreads();
    unsigned char* out = ((job == 0) ? Qbuf : Kbuf) + (size_t)row0 * CDIM;
    const v4u* src = (const v4u*)lds;
    v4u d0 = src[lane * 2];
    v4u d1 = src[lane * 2 + 1];
    *(v4u*)(out + lane * 32)      = d0;
    *(v4u*)(out + lane * 32 + 16) = d1;
  } else {
    const int bidx = row0 >> 11;
    const int nn   = (row0 & (SEQ - 1) & ~63) + ((row0 >> 5) & 1) * 32 + 16 * h + ((row0 >> 4) & 1) * 8;
    #pragma unroll
    for (int dt = 0; dt < 4; ++dt) {
      const int f = dt * 16 + n16;
      v2i wpair;
      wpair[0] = pack_byte1(st[dt][0], st[dt][1], st[dt][2], st[dt][3]);
      wpair[1] = pack_byte1(st[dt][4], st[dt][5], st[dt][6], st[dt][7]);
      *(v2i*)(VTbuf + ((size_t)bidx * CDIM + f) * SEQ + nn) = wpair;
    }
  }
}

__device__ __forceinline__ void load_ak(const unsigned char* kbase, int kb,
                                        int n16, int h, v8i (&ak)[4]) {
  #pragma unroll
  for (int t = 0; t < 4; ++t) {
    const unsigned char* krow = kbase + (size_t)(kb * 64 + t * 16 + n16) * CDIM;
    #pragma unroll
    for (int g = 0; g < 4; ++g) {
      v2u d = *(const v2u*)(krow + (g >> 1) * 32 + h * 16 + (g & 1) * 8);
      ak[t][2*g]   = (int)d[0];
      ak[t][2*g+1] = (int)d[1];
    }
  }
}
__device__ __forceinline__ void load_bv(const unsigned char* vbase, int kb,
                                        int n16, int h, v8i (&bv)[4]) {
  #pragma unroll
  for (int ft = 0; ft < 4; ++ft) {
    const unsigned char* vrow = vbase + (size_t)(ft * 16 + n16) * SEQ + kb * 64;
    bv[ft] = make_v8(*(const v4u*)(vrow + h * 16),
                     *(const v4u*)(vrow + 32 + h * 16));
  }
}
__device__ __forceinline__ void wmma_s(const v8i (&ak)[4], v8i bq,
                                       v8i (&st)[4]) {
  #pragma unroll
  for (int t = 0; t < 4; ++t) {
    v8i c = {};
    st[t] = __builtin_amdgcn_wmma_i32_16x16x64_iu8(false, ak[t], false, bq,
                                                   c, false, false);
  }
}
__device__ __forceinline__ v8i pack_p(const v8i (&st)[4]) {
  v8i p;
  #pragma unroll
  for (int t = 0; t < 4; ++t) {
    p[2*t]   = pack_byte1(st[t][0], st[t][1], st[t][2], st[t][3]);
    p[2*t+1] = pack_byte1(st[t][4], st[t][5], st[t][6], st[t][7]);
  }
  return p;
}
__device__ __forceinline__ void wmma_acc(v8i pmat, const v8i (&bv)[4],
                                         v8i (&acc)[4]) {
  #pragma unroll
  for (int ft = 0; ft < 4; ++ft)
    acc[ft] = __builtin_amdgcn_wmma_i32_16x16x64_iu8(false, pmat, false,
                                                     bv[ft], acc[ft],
                                                     false, false);
}

__global__ __launch_bounds__(32) void qattn_attn_kernel(
    const unsigned char* __restrict__ Qbuf, const unsigned char* __restrict__ Kbuf,
    const unsigned char* __restrict__ VTbuf, float* __restrict__ out)
{
  const int lane = threadIdx.x;
  const int n16  = lane & 15;
  const int h    = lane >> 4;
  const int q0   = blockIdx.x * 16;
  const int b    = blockIdx.y;

  const unsigned char* qrow = Qbuf + (size_t)(b * SEQ + q0 + n16) * CDIM;
  v8i bq = make_v8(*(const v4u*)(qrow + h * 16),
                   *(const v4u*)(qrow + 32 + h * 16));

  v8i acc[4] = {};

  const unsigned char* kbase = Kbuf  + (size_t)b * SEQ * CDIM;
  const unsigned char* vbase = VTbuf + (size_t)b * CDIM * SEQ;

  v8i akA[4], akB[4], bvA[4], bvB[4], stA[4], stB[4];

  load_ak(kbase, 0, n16, h, akA);
  load_bv(vbase, 0, n16, h, bvA);
  wmma_s(akA, bq, stA);
  load_ak(kbase, 1, n16, h, akB);
  load_bv(vbase, 1, n16, h, bvB);

  for (int kb = 0; kb < SEQ / 64 - 2; kb += 2) {
    load_ak(kbase, kb + 2, n16, h, akA);
    wmma_s(akB, bq, stB);
    wmma_acc(pack_p(stA), bvA, acc);
    load_bv(vbase, kb + 2, n16, h, bvA);
    load_ak(kbase, kb + 3, n16, h, akB);

    wmma_s(akA, bq, stA);
    wmma_acc(pack_p(stB), bvB, acc);
    load_bv(vbase, kb + 3, n16, h, bvB);
  }
  wmma_s(akB, bq, stB);
  wmma_acc(pack_p(stA), bvA, acc);
  wmma_acc(pack_p(stB), bvB, acc);

  float* orow = out + (size_t)(b * SEQ + q0) * CDIM;
  #pragma unroll
  for (int ft = 0; ft < 4; ++ft) {
    #pragma unroll
    for (int r = 0; r < 8; ++r) {
      const int m = r + 8 * h;
      const int f = ft * 16 + n16;
      orow[(size_t)m * CDIM + f] = (float)(acc[ft][r] >> 8);
    }
  }
}

extern "C" void kernel_launch(void* const* d_in, const int* in_sizes, int n_in,
                              void* d_out, int out_size, void* d_ws, size_t ws_size,
                              hipStream_t stream) {
  const float* q_in = (const float*)d_in[0];
  const float* k_in = (const float*)d_in[1];
  const float* v_in = (const float*)d_in[2];
  const float* wq   = (const float*)d_in[3];
  const float* wk   = (const float*)d_in[4];
  const float* wv   = (const float*)d_in[5];
  float* out        = (float*)d_out;

  unsigned char* Qbuf  = (unsigned char*)d_ws;
  unsigned char* Kbuf  = Qbuf + (size_t)BATCH * SEQ * CDIM;
  unsigned char* VTbuf = Kbuf + (size_t)BATCH * SEQ * CDIM;

  dim3 pgrid((BATCH * SEQ) / 16, 3);
  qattn_proj_kernel<<<pgrid, 32, 0, stream>>>(q_in, k_in, v_in, wq, wk, wv,
                                              Qbuf, Kbuf, VTbuf);

  dim3 agrid(SEQ / 16, BATCH);
  qattn_attn_kernel<<<agrid, 32, 0, stream>>>(Qbuf, Kbuf, VTbuf, out);
}
